// Hopfield_3839700763245
// MI455X (gfx1250) — matrix operand order repaired and checked by reading, not compiled, not run on hardware
//
#include <hip/hip_runtime.h>
#include <hip/hip_bf16.h>

#define LATENT 4096
#define LABEL  512
#define NBLK   (LATENT / 16)   // 256 blocks of 16 columns

typedef __attribute__((ext_vector_type(16))) __bf16 v16bf;
typedef __attribute__((ext_vector_type(8)))  float  v8f;

static __device__ __forceinline__ float bf2f(unsigned short u) {
    union { unsigned int i; float f; } c;
    c.i = ((unsigned int)u) << 16;
    return c.f;
}
static __device__ __forceinline__ __bf16 us2bf(unsigned short u) {
    union { unsigned short u; __bf16 b; } c;
    c.u = u;
    return c.b;
}

// ---------------------------------------------------------------------------
// Kernel 0: zero the sign-sum accumulator (workspace is poisoned by harness)
// ---------------------------------------------------------------------------
__global__ void k_init(float* signsum) {
    if (threadIdx.x == 0) signsum[0] = 0.0f;
}

// ---------------------------------------------------------------------------
// Kernel 1: x = sign(clip(llv)) in {±1}, stored as bf16 bits TRANSPOSED:
//           xT[i*512 + k] = sign(llv[k*4096 + i]).  Also exact sum of signs.
// ---------------------------------------------------------------------------
__global__ void k_sign_transpose(const float* __restrict__ llv,
                                 unsigned short* __restrict__ xT,
                                 float* __restrict__ signsum) {
    __shared__ unsigned short tile[16][17];
    __shared__ int cnt;
    const int tx = threadIdx.x, ty = threadIdx.y;
    if (tx == 0 && ty == 0) cnt = 0;
    __syncthreads();

    const int i0 = blockIdx.x * 16;   // latent-dim block
    const int k0 = blockIdx.y * 16;   // label-dim block
    const float v = llv[(size_t)(k0 + ty) * LATENT + (i0 + tx)];
    const bool pos = (v >= 0.0f);     // sign(clip(v)): clip preserves sign; sign(0)=+1
    tile[ty][tx] = pos ? (unsigned short)0x3F80 : (unsigned short)0xBF80;
    atomicAdd(&cnt, pos ? 1 : -1);
    __syncthreads();

    // coalesced transposed store along k
    xT[(size_t)(i0 + ty) * LABEL + (k0 + tx)] = tile[tx][ty];
    if (tx == 0 && ty == 0) atomicAdd(signsum, (float)cnt);  // integer-exact
}

// ---------------------------------------------------------------------------
// Kernel 2: per-16-column Gram tiles of x (block diagonal of x^T x) via WMMA.
//   Graw[b][j][j'] = sum_k x[k, b*16+j] * x[k, b*16+j']   (exact integers)
//   colsum[i]      = sum_k x[k, i]
// One wave (32 threads) per block b; K-loop of 16 v_wmma_f32_16x16x32_bf16.
// ---------------------------------------------------------------------------
__global__ void __launch_bounds__(32) k_gram(const unsigned short* __restrict__ xT,
                                             float* __restrict__ Graw,
                                             float* __restrict__ colsum) {
    __shared__ unsigned short tile[16 * LABEL];   // 16 rows x 512, 16 KB
    const int lane = threadIdx.x;                 // 0..31
    const int b = blockIdx.x;                     // 0..255
    const int i0 = b * 16;

    // rows i0..i0+15 of xT are one contiguous 16 KB chunk: bulk 128-bit copy
    {
        const uint4* src = (const uint4*)(xT + (size_t)i0 * LABEL);
        uint4* dst = (uint4*)tile;
        #pragma unroll
        for (int it = 0; it < 32; ++it) dst[it * 32 + lane] = src[it * 32 + lane];
    }
    __syncthreads();

    const int m = lane & 15;     // matrix row (A) / column (B)
    const int h = lane >> 4;     // half select
    const unsigned short* rowm = tile + m * LABEL;

    v8f c = {};
    #pragma unroll
    for (int kk0 = 0; kk0 < LABEL; kk0 += 32) {
        // A (16x32, bf16): lanes 0-15 hold M=m, K in {0..7, 16..23};
        //                  lanes 16-31 hold M=m, K in {8..15, 24..31}
        v16bf a, bb;
        #pragma unroll
        for (int t = 0; t < 8; ++t)  a[t]     = us2bf(rowm[kk0 + h * 8 + t]);
        #pragma unroll
        for (int t = 0; t < 8; ++t)  a[8 + t] = us2bf(rowm[kk0 + 16 + h * 8 + t]);
        // B (32x16, bf16): column m, same K layout as A
        #pragma unroll
        for (int t = 0; t < 8; ++t)  bb[t]     = us2bf(rowm[kk0 + h * 8 + t]);
        #pragma unroll
        for (int t = 0; t < 8; ++t)  bb[8 + t] = us2bf(rowm[kk0 + 16 + h * 8 + t]);
        c = __builtin_amdgcn_wmma_f32_16x16x32_bf16(false, a, false, bb,
                                                    (short)0, c, false, false);
    }

    // C/D layout: lane<16 -> rows 0..7 (vgpr r), lane>=16 -> rows 8..15; col = m
    float* gout = Graw + (size_t)b * 256;
    #pragma unroll
    for (int r = 0; r < 8; ++r) gout[(r + 8 * h) * 16 + m] = c[r];

    // column sums (exact): row j=m, half h covers 256 elements
    float sum = 0.0f;
    for (int k = h * 256; k < h * 256 + 256; ++k) sum += bf2f(rowm[k]);
    sum += __shfl_xor(sum, 16, 32);
    if (h == 0) colsum[i0 + m] = sum;
}

// ---------------------------------------------------------------------------
// Kernel 3: blocked sequential Gauss-Seidel sweep + scores + argmax one-hot.
// Single workgroup of 512 threads (16 waves); thread k owns d_k.
// ---------------------------------------------------------------------------
__global__ void __launch_bounds__(512) k_sweep(const float* __restrict__ s_in,
                                               const unsigned short* __restrict__ xT,
                                               const float* __restrict__ Graw,
                                               const float* __restrict__ colsum,
                                               const float* __restrict__ signsum,
                                               float* __restrict__ out) {
    __shared__ float s_lds[LATENT];   // 16 KB, updated in place
    __shared__ float red[256];        // per-wave reduction staging
    __shared__ float u[16];
    __shared__ float Gt[256];
    __shared__ float delta[16];
    __shared__ float ssum_sh;
    __shared__ float sc[LABEL];
    __shared__ int   idxs[LABEL];

    const int k = threadIdx.x;        // 0..511
    const int wave = k >> 5, lane = k & 31;

    for (int i = k; i < LATENT; i += LABEL) s_lds[i] = s_in[i];
    __syncthreads();

    // ssum = sum(s)
    {
        float ps = 0.0f;
        for (int i = k; i < LATENT; i += LABEL) ps += s_lds[i];
        #pragma unroll
        for (int off = 16; off; off >>= 1) ps += __shfl_xor(ps, off, 32);
        if (lane == 0) red[wave] = ps;
        __syncthreads();
        if (k == 0) {
            float t = 0.0f;
            for (int w = 0; w < 16; ++w) t += red[w];
            ssum_sh = t;
        }
        __syncthreads();
    }
    const float rho = signsum[0] / (float)(LATENT * LABEL);
    const float ssum = ssum_sh;

    // Phase A: d_k = sum_i temp[k,i]*s[i] = sum_i x[k,i]*s[i] - rho*sum(s)
    float d = 0.0f;
    for (int i = 0; i < LATENT; ++i)
        d += bf2f(xT[(size_t)i * LABEL + k]) * s_lds[i];
    d -= rho * ssum;

    // Phase B: 256 blocks of 16 sequential updates
    const float rr = (float)LABEL * rho * rho;
    for (int b = 0; b < NBLK; ++b) {
        const int i0 = b * 16;
        float t[16];
        #pragma unroll
        for (int j = 0; j < 16; ++j)
            t[j] = bf2f(xT[(size_t)(i0 + j) * LABEL + k]) - rho;   // coalesced

        // Gram of temp columns: Gt = Graw - rho*(cs_j + cs_j') + 512*rho^2
        if (k < 256) {
            const int jp = k >> 4, jj = k & 15;
            Gt[k] = Graw[(size_t)b * 256 + k]
                    - rho * (colsum[i0 + jp] + colsum[i0 + jj]) + rr;
        }

        // u_j = sum_k t[j]*d_k  (block-start d)
        #pragma unroll
        for (int j = 0; j < 16; ++j) {
            float p = t[j] * d;
            #pragma unroll
            for (int off = 16; off; off >>= 1) p += __shfl_xor(p, off, 32);
            if (lane == 0) red[wave * 16 + j] = p;
        }
        __syncthreads();
        if (k < 16) {
            float uj = 0.0f;
            for (int w = 0; w < 16; ++w) uj += red[w * 16 + k];
            u[k] = uj;
        }
        __syncthreads();

        // 16-step sequential recurrence (scalar)
        if (k == 0) {
            #pragma unroll 1
            for (int jj = 0; jj < 16; ++jj) {
                float v = u[jj];
                for (int jp = 0; jp < jj; ++jp) v += delta[jp] * Gt[jp * 16 + jj];
                const float sold = s_lds[i0 + jj];
                v -= sold * Gt[jj * 16 + jj];              // zeroed diagonal of W
                const float snew = (v >= 0.0f) ? 1.0f : -1.0f;
                delta[jj] = snew - sold;
                s_lds[i0 + jj] = snew;
            }
        }
        __syncthreads();

        // rank-16 update of d
        #pragma unroll
        for (int j = 0; j < 16; ++j) d += delta[j] * t[j];
        __syncthreads();   // protect delta/red/Gt reuse next block
    }

    // Phase C: scores[k] = |sum_i x[k,i]*s[i]| / 4096   (exact: all ±1)
    float acc = 0.0f;
    for (int i = 0; i < LATENT; ++i)
        acc += bf2f(xT[(size_t)i * LABEL + k]) * s_lds[i];
    const float score = fabsf(acc) / (float)LATENT;
    sc[k] = score;
    idxs[k] = k;
    __syncthreads();

    // Phase D: first-index argmax, one-hot output
    for (int stride = 256; stride; stride >>= 1) {
        if (k < stride) {
            const float a = sc[k], bv = sc[k + stride];
            const int ia = idxs[k], ib = idxs[k + stride];
            if (bv > a || (bv == a && ib < ia)) { sc[k] = bv; idxs[k] = ib; }
        }
        __syncthreads();
    }
    const int amax = idxs[0];
    out[k] = (k == amax) ? score : 0.0f;
}

// ---------------------------------------------------------------------------
extern "C" void kernel_launch(void* const* d_in, const int* in_sizes, int n_in,
                              void* d_out, int out_size, void* d_ws, size_t ws_size,
                              hipStream_t stream) {
    (void)in_sizes; (void)n_in; (void)out_size; (void)ws_size;
    const float* s   = (const float*)d_in[0];   // [4096]
    const float* llv = (const float*)d_in[1];   // [512, 4096]

    char* ws = (char*)d_ws;
    unsigned short* xT   = (unsigned short*)ws;                              // 4 MiB
    float*          Graw = (float*)(ws + (size_t)LATENT * LABEL * 2);        // 256 KiB
    float*          csum = (float*)((char*)Graw + (size_t)NBLK * 256 * 4);   // 16 KiB
    float*          ssum = (float*)((char*)csum + (size_t)LATENT * 4);       // 4 B

    hipLaunchKernelGGL(k_init, dim3(1), dim3(1), 0, stream, ssum);
    hipLaunchKernelGGL(k_sign_transpose, dim3(LATENT / 16, LABEL / 16),
                       dim3(16, 16), 0, stream, llv, xT, ssum);
    hipLaunchKernelGGL(k_gram, dim3(NBLK), dim3(32), 0, stream, xT, Graw, csum);
    hipLaunchKernelGGL(k_sweep, dim3(1), dim3(512), 0, stream,
                       s, xT, Graw, csum, ssum, (float*)d_out);
}
